// ASA_21466246545927
// MI455X (gfx1250) — matrix operand order repaired and checked by reading, not compiled, not run on hardware
//
#include <hip/hip_runtime.h>
#include <hip/hip_bf16.h>

// ---------------------------------------------------------------------------
// ASA block on MI455X (gfx1250, wave32, WMMA bf16 16x16x32 w/ f32 accumulate)
// B=4 C=64 F=256 T=400 d_c=16
// ---------------------------------------------------------------------------

typedef __attribute__((ext_vector_type(16))) __bf16 v16bf;
typedef __attribute__((ext_vector_type(8)))  __bf16 v8bf;
typedef __attribute__((ext_vector_type(8)))  float  v8f;

#define BB 4
#define CC 64
#define FF 256
#define TT 400
#define DD 16

__device__ __forceinline__ v8f wmma_bf16(v16bf a, v16bf b, v8f c) {
  return __builtin_amdgcn_wmma_f32_16x16x32_bf16(false, a, false, b, (short)0, c,
                                                 false, false);
}

// Branchless PReLU: y>0 ? y : a*y
__device__ __forceinline__ float prelu(float y, float a) {
  return fmaxf(y, 0.f) + a * fminf(y, 0.f);
}

// A-matrix (16x32 bf16) per-lane element->K map: K = e + 16*(lane>=16)
__device__ __forceinline__ int afk(int lane, int e) {
  return e + ((lane >> 4) << 4);
}
// B-matrix (32x16 bf16): n = lane&15, K = e + 16*(lane>=16)
__device__ __forceinline__ int bfk(int lane, int e) {
  return e + ((lane >> 4) << 4);
}

// Fast fragment loads when the source row is K-contiguous in LDS.
__device__ __forceinline__ v16bf load_a_row(const __bf16* row, int k0, int lane) {
  return *(const v16bf*)(row + k0 + ((lane >> 4) << 4));
}
__device__ __forceinline__ v16bf load_b_row(const __bf16* row, int k0, int lane) {
  return *(const v16bf*)(row + k0 + ((lane >> 4) << 4));
}

// ---------------------------------------------------------------------------
// Kernel 1: fused conv1x1 (80 output rows: 48 fqkv + 32 tqk) + BN + PReLU.
// Block handles (b, f0..f0+31, t0..t0+15). Tiles of 16 consecutive t keep inp
// reads and qt/kt writes coalesced; qf/kf/vf are transposed through LDS so
// their [b][t][c][f] stores are 64B-contiguous.
// ---------------------------------------------------------------------------
__global__ void __launch_bounds__(512) k_qkv(
    const float* __restrict__ x, const float* __restrict__ w_fqkv,
    const float* __restrict__ w_tqk,
    const float* __restrict__ g1, const float* __restrict__ b1,
    const float* __restrict__ m1, const float* __restrict__ v1,
    const float* __restrict__ a1,
    const float* __restrict__ g2, const float* __restrict__ b2,
    const float* __restrict__ m2, const float* __restrict__ v2,
    const float* __restrict__ a2,
    __bf16* __restrict__ qf, __bf16* __restrict__ kf, __bf16* __restrict__ vf,
    __bf16* __restrict__ qt, __bf16* __restrict__ kt) {
  __shared__ float bnsc[80], bnsh[80], bnal[80];
  __shared__ __bf16 stage[16 * 48 * 32];  // [t_loc][c48][f_loc]
  int tid = threadIdx.x;
  if (tid < 80) {
    float g, bbv, mmv, vvv, aav;
    if (tid < 48) { g = g1[tid]; bbv = b1[tid]; mmv = m1[tid]; vvv = v1[tid]; aav = a1[tid]; }
    else { int c = tid - 48; g = g2[c]; bbv = b2[c]; mmv = m2[c]; vvv = v2[c]; aav = a2[c]; }
    float inv = rsqrtf(vvv + 1e-5f);
    bnsc[tid] = g * inv;
    bnsh[tid] = bbv - g * mmv * inv;
    bnal[tid] = aav;
  }
  __syncthreads();

  int bI = blockIdx.x;
  int t0 = (bI % 25) * 16; bI /= 25;
  int f0 = (bI % 8) * 32;
  int b  = bI / 8;
  int lane = tid & 31, w = tid >> 5;
  int m = lane & 15;

  // W fragments: 5 row-tiles x 2 k-steps, converted fp32->bf16 on the fly.
  v16bf aw[5][2];
#pragma unroll
  for (int r = 0; r < 5; ++r)
#pragma unroll
    for (int ks = 0; ks < 2; ++ks)
#pragma unroll
      for (int e = 0; e < 16; ++e) {
        int k = ks * 32 + afk(lane, e);
        int row = r * 16 + m;
        float wv = (row < 48) ? w_fqkv[row * 64 + k] : w_tqk[(row - 48) * 64 + k];
        aw[r][ks][e] = (__bf16)wv;
      }

  for (int j = 0; j < 4; ++j) {
    int fl = w * 4 + j, f = f0 + fl;
    int n = lane & 15;
    v16bf bx[2];
#pragma unroll
    for (int ks = 0; ks < 2; ++ks)
#pragma unroll
      for (int e = 0; e < 16; ++e) {
        int k = ks * 32 + bfk(lane, e);
        bx[ks][e] = (__bf16)x[((b * CC + k) * FF + f) * TT + t0 + n];
      }
#pragma unroll
    for (int r = 0; r < 5; ++r) {
      v8f acc = {};
      acc = wmma_bf16(aw[r][0], bx[0], acc);
      acc = wmma_bf16(aw[r][1], bx[1], acc);
#pragma unroll
      for (int i = 0; i < 8; ++i) {
        int mr = i + ((lane >> 4) << 3);
        int ch = r * 16 + mr;
        float y = prelu(acc[i] * bnsc[ch] + bnsh[ch], bnal[ch]);
        if (r < 3) {
          stage[(n * 48 + ch) * 32 + fl] = (__bf16)y;
        } else {
          __bf16* dst = (r == 3) ? qt : kt;  // uniform in unrolled r
          dst[((b * FF + f) * DD + (mr & 15)) * TT + t0 + n] = (__bf16)y;
        }
      }
    }
  }
  __syncthreads();
  // coalesced write-out of qf/kf/vf: 16t x 48c rows of 32 f (64B each)
  for (int i = tid; i < 16 * 48 * 4; i += 512) {
    int q = i & 3; int rc = i >> 2;
    int c48 = rc % 48; int tl = rc / 48;
    __bf16* buf = (c48 < 16) ? qf : (c48 < 32 ? kf : vf);
    uint4 v4 = *reinterpret_cast<const uint4*>(&stage[(tl * 48 + c48) * 32 + q * 8]);
    *reinterpret_cast<uint4*>(
        &buf[((size_t)(b * TT + t0 + tl) * DD + (c48 & 15)) * FF + f0 + q * 8]) = v4;
  }
}

// ---------------------------------------------------------------------------
// Kernel 2: F-attention, one block per (b,t). 8 waves, 2 f-tiles each.
// ---------------------------------------------------------------------------
__global__ void __launch_bounds__(256) k_fattn(
    const __bf16* __restrict__ qf, const __bf16* __restrict__ kf,
    const __bf16* __restrict__ vf, __bf16* __restrict__ fo) {
  extern __shared__ char smem[];
  __bf16* qs  = (__bf16*)smem;          // 16 x 256
  __bf16* ksm = qs + DD * FF;
  __bf16* vs  = ksm + DD * FF;
  float*  S   = (float*)(smem + 3 * DD * FF * 2);          // 8 waves x 16x256 f32
  __bf16* P   = (__bf16*)(smem + 3 * DD * FF * 2 + 8 * DD * FF * 4);
  int tid = threadIdx.x, lane = tid & 31, w = tid >> 5;
  int t = blockIdx.x % TT, b = blockIdx.x / TT;
  size_t base = ((size_t)(b * TT + t) * DD) * FF;

  const uint4* s0 = (const uint4*)(qf + base);
  const uint4* s1 = (const uint4*)(kf + base);
  const uint4* s2 = (const uint4*)(vf + base);
  uint4* d0 = (uint4*)qs; uint4* d1 = (uint4*)ksm; uint4* d2 = (uint4*)vs;
  for (int i = tid; i < 512; i += 256) { d0[i] = s0[i]; d1[i] = s1[i]; d2[i] = s2[i]; }
  __syncthreads();

  float*  Sw = S + w * DD * FF;
  __bf16* Pw = P + w * DD * FF;
  int m = lane & 15;

  for (int ft = w; ft < 16; ft += 8) {
    int f0 = ft * 16;
    // A[m,k] = qf[k][f0+m], K=16 padded to 32 with zeros
    v16bf A;
#pragma unroll
    for (int e = 0; e < 16; ++e) {
      int k = afk(lane, e);
      A[e] = (k < DD) ? qs[k * FF + f0 + m] : (__bf16)0.0f;
    }
    for (int yt = 0; yt < 16; ++yt) {
      int y0 = yt * 16;
      v16bf Bf;
#pragma unroll
      for (int e = 0; e < 16; ++e) {
        int k = bfk(lane, e);
        Bf[e] = (k < DD) ? ksm[k * FF + y0 + m] : (__bf16)0.0f;
      }
      v8f Dv = {};
      Dv = wmma_bf16(A, Bf, Dv);
#pragma unroll
      for (int i = 0; i < 8; ++i)
        Sw[(i + ((lane >> 4) << 3)) * FF + y0 + m] = Dv[i] * 0.25f;
    }
    __threadfence_block();
    // row softmax (256 wide)
    for (int r = 0; r < 16; ++r) {
      float mx = -3.0e38f;
      for (int y = lane; y < FF; y += 32) mx = fmaxf(mx, Sw[r * FF + y]);
      for (int o = 16; o > 0; o >>= 1) mx = fmaxf(mx, __shfl_xor(mx, o, 32));
      float sum = 0.f;
      for (int y = lane; y < FF; y += 32) {
        float e2 = __expf(Sw[r * FF + y] - mx);
        Sw[r * FF + y] = e2; sum += e2;
      }
      for (int o = 16; o > 0; o >>= 1) sum += __shfl_xor(sum, o, 32);
      float inv = 1.0f / sum;
      for (int y = lane; y < FF; y += 32) Pw[r * FF + y] = (__bf16)(Sw[r * FF + y] * inv);
    }
    __threadfence_block();
    // f_out[c][f_loc] = sum_y v[c,y] * P[f_loc,y]   (K=256, 8 steps)
    v8f acc = {};
    for (int k0 = 0; k0 < FF; k0 += 32) {
      v16bf A2 = load_a_row(vs + m * FF, k0, lane);
      v16bf B2 = load_b_row(Pw + m * FF, k0, lane);
      acc = wmma_bf16(A2, B2, acc);
    }
#pragma unroll
    for (int i = 0; i < 8; ++i) {
      int c = i + ((lane >> 4) << 3);
      fo[base + (size_t)c * FF + f0 + m] = (__bf16)acc[i];
    }
    __threadfence_block();
  }
}

// ---------------------------------------------------------------------------
// Kernel 3: transpose f_out [b][t][c][f] -> [b][f][c][t] (16x16 LDS tiles)
// ---------------------------------------------------------------------------
__global__ void __launch_bounds__(256) k_tr(const __bf16* __restrict__ fo,
                                            __bf16* __restrict__ foT) {
  __shared__ __bf16 tile[16][17];
  int bi = blockIdx.x;
  int tt = bi % 25; bi /= 25;
  int ff = bi % 16; bi /= 16;
  int c  = bi % 16; int b = bi / 16;
  int tx = threadIdx.x & 15, ty = threadIdx.x >> 4;
  tile[ty][tx] = fo[((size_t)(b * TT + tt * 16 + ty) * DD + c) * FF + ff * 16 + tx];
  __syncthreads();
  foT[((size_t)(b * FF + ff * 16 + ty) * DD + c) * TT + tt * 16 + tx] = tile[tx][ty];
}

// ---------------------------------------------------------------------------
// Kernel 4: causal T-attention + proj conv + BN + PReLU + residual.
// One block per (b,f), 6 waves; causality via computing only y-tiles <= t-tile
// and softmax over y <= t (so the triu mask never needs materializing).
// ---------------------------------------------------------------------------
#define W5 6
#define WD 416  // 400 padded to 13*32, zero-filled
__global__ void __launch_bounds__(192) k_tattn(
    const __bf16* __restrict__ qt, const __bf16* __restrict__ kt,
    const __bf16* __restrict__ foT, const float* __restrict__ x,
    const float* __restrict__ wproj,
    const float* __restrict__ g3, const float* __restrict__ b3,
    const float* __restrict__ m3, const float* __restrict__ v3,
    const float* __restrict__ a3, float* __restrict__ out) {
  extern __shared__ char smem[];
  __bf16* qs  = (__bf16*)smem;                   // 16 x 416
  __bf16* ksm = qs + DD * WD;
  __bf16* fs  = ksm + DD * WD;
  float* wp   = (float*)(smem + 3 * DD * WD * 2);  // 64x16
  float* sc3  = wp + 1024; float* sh3 = sc3 + 64; float* al3 = sh3 + 64;
  float* tout = al3 + 64;                          // 6 waves x 16x16
  float* S    = tout + W5 * 256;                   // 6 waves x 16x416 f32
  __bf16* P   = (__bf16*)(S + W5 * DD * WD);       // 6 waves x 16x416 bf16

  int tid = threadIdx.x, lane = tid & 31, w = tid >> 5;
  int f = blockIdx.x % FF, b = blockIdx.x / FF;
  size_t base = ((size_t)(b * FF + f) * DD) * TT;

  for (int i = tid; i < 3 * 16 * 50; i += 192) {
    int bu = i / 800; int rem = i % 800; int r = rem / 50; int q = rem % 50;
    const __bf16* sp = (bu == 0 ? qt : (bu == 1 ? kt : foT)) + base + r * TT + q * 8;
    __bf16* dp = (bu == 0 ? qs : (bu == 1 ? ksm : fs)) + r * WD + q * 8;
    *(uint4*)dp = *(const uint4*)sp;
  }
  for (int i = tid; i < 3 * 16 * 16; i += 192) {  // zero the K padding
    int bu = i / 256; int rem = i % 256; int r = rem / 16;
    (bu == 0 ? qs : (bu == 1 ? ksm : fs))[r * WD + 400 + (rem % 16)] = (__bf16)0.0f;
  }
  for (int i = tid; i < 1024; i += 192) wp[i] = wproj[i];
  if (tid < 64) {
    float inv = rsqrtf(v3[tid] + 1e-5f);
    sc3[tid] = g3[tid] * inv;
    sh3[tid] = b3[tid] - g3[tid] * m3[tid] * inv;
    al3[tid] = a3[tid];
  }
  __syncthreads();

  float*  Sw = S + w * DD * WD;
  __bf16* Pw = P + w * DD * WD;
  float*  tw = tout + w * 256;
  int m = lane & 15;

  for (int tt = w; tt < 25; tt += W5) {
    int t0 = tt * 16;
    v16bf A;
#pragma unroll
    for (int e = 0; e < 16; ++e) {
      int k = afk(lane, e);
      A[e] = (k < DD) ? qs[k * WD + t0 + m] : (__bf16)0.0f;
    }
    for (int yt = 0; yt <= tt; ++yt) {  // causal: skip fully-masked tiles
      int y0 = yt * 16;
      v16bf Bf;
#pragma unroll
      for (int e = 0; e < 16; ++e) {
        int k = bfk(lane, e);
        Bf[e] = (k < DD) ? ksm[k * WD + y0 + m] : (__bf16)0.0f;
      }
      v8f Dv = {};
      Dv = wmma_bf16(A, Bf, Dv);
#pragma unroll
      for (int i = 0; i < 8; ++i)
        Sw[(i + ((lane >> 4) << 3)) * WD + y0 + m] = Dv[i] * 0.25f;
    }
    __threadfence_block();
    for (int r = 0; r < 16; ++r) {
      int nv = t0 + r + 1;  // valid y range: y <= t
      float mx = -3.0e38f;
      for (int y = lane; y < nv; y += 32) mx = fmaxf(mx, Sw[r * WD + y]);
      for (int o = 16; o > 0; o >>= 1) mx = fmaxf(mx, __shfl_xor(mx, o, 32));
      float sum = 0.f;
      for (int y = lane; y < nv; y += 32) {
        float e2 = __expf(Sw[r * WD + y] - mx);
        Sw[r * WD + y] = e2; sum += e2;
      }
      for (int o = 16; o > 0; o >>= 1) sum += __shfl_xor(sum, o, 32);
      float inv = 1.0f / sum;
      for (int y = lane; y < WD; y += 32)
        Pw[r * WD + y] = (y < nv) ? (__bf16)(Sw[r * WD + y] * inv) : (__bf16)0.0f;
    }
    __threadfence_block();
    // t_out[c][t_loc] = sum_y foT[c,y] * P[t_loc,y]   (K=416 padded, 13 steps)
    v8f acc = {};
    for (int k0 = 0; k0 < WD; k0 += 32) {
      v16bf A2 = load_a_row(fs + m * WD, k0, lane);
      v16bf B2 = load_b_row(Pw + m * WD, k0, lane);
      acc = wmma_bf16(A2, B2, acc);
    }
#pragma unroll
    for (int i = 0; i < 8; ++i)
      tw[(i + ((lane >> 4) << 3)) * 16 + m] = acc[i];
    __threadfence_block();
    // proj (64x16) + BN + PReLU + residual, 64 ch x 16 t per tile
#pragma unroll
    for (int oo = 0; oo < 2; ++oo) {
      int o = lane + oo * 32;
      float sc = sc3[o], sh = sh3[o], al = al3[o];
      for (int tl = 0; tl < 16; ++tl) {
        float a2 = 0.f;
#pragma unroll
        for (int c = 0; c < 16; ++c) a2 += wp[o * 16 + c] * tw[c * 16 + tl];
        float y = prelu(a2 * sc + sh, al);
        size_t idx = ((size_t)(b * CC + o) * FF + f) * TT + t0 + tl;
        out[idx] = y + x[idx];
      }
    }
    __threadfence_block();
  }
}

// ---------------------------------------------------------------------------
extern "C" void kernel_launch(void* const* d_in, const int* in_sizes, int n_in,
                              void* d_out, int out_size, void* d_ws, size_t ws_size,
                              hipStream_t stream) {
  (void)in_sizes; (void)n_in; (void)out_size; (void)ws_size;
  const float* inp    = (const float*)d_in[0];
  // d_in[1] = mask (implicit causal; not needed)
  const float* w_fqkv = (const float*)d_in[2];
  const float* g1 = (const float*)d_in[3];
  const float* b1 = (const float*)d_in[4];
  const float* m1 = (const float*)d_in[5];
  const float* v1 = (const float*)d_in[6];
  const float* a1 = (const float*)d_in[7];
  const float* w_tqk = (const float*)d_in[8];
  const float* g2 = (const float*)d_in[9];
  const float* b2 = (const float*)d_in[10];
  const float* m2 = (const float*)d_in[11];
  const float* v2 = (const float*)d_in[12];
  const float* a2 = (const float*)d_in[13];
  const float* w_proj = (const float*)d_in[14];
  const float* g3 = (const float*)d_in[15];
  const float* b3 = (const float*)d_in[16];
  const float* m3 = (const float*)d_in[17];
  const float* v3 = (const float*)d_in[18];
  const float* a3 = (const float*)d_in[19];

  const size_t SZ = (size_t)BB * TT * DD * FF;  // 6,553,600 elems (bf16)
  __bf16* ws = (__bf16*)d_ws;
  __bf16 *qf = ws, *kf = ws + SZ, *vf = ws + 2 * SZ;
  __bf16 *qt = ws + 3 * SZ, *kt = ws + 4 * SZ;
  __bf16 *fo = ws + 5 * SZ, *foT = ws + 6 * SZ;

  // Stage 1: fused conv1x1 (fqkv + tqk) + BN + PReLU, bf16 layouts for attn
  k_qkv<<<800, 512, 0, stream>>>(inp, w_fqkv, w_tqk, g1, b1, m1, v1, a1,
                                 g2, b2, m2, v2, a2, qf, kf, vf, qt, kt);
  // Stage 2: F-attention per (b,t)
  size_t sh2 = (size_t)3 * DD * FF * 2 + (size_t)8 * DD * FF * 4 + (size_t)8 * DD * FF * 2;
  k_fattn<<<1600, 256, sh2, stream>>>(qf, kf, vf, fo);
  // Stage 3: transpose f_out into [b][f][c][t]
  k_tr<<<25600, 256, 0, stream>>>(fo, foT);
  // Stage 4: causal T-attention + proj + BN + PReLU + residual
  size_t sh4 = (size_t)3 * DD * WD * 2 + 1024 * 4 + 3 * 64 * 4 + (size_t)W5 * 256 * 4
             + (size_t)W5 * DD * WD * 4 + (size_t)W5 * DD * WD * 2;
  k_tattn<<<1024, 192, sh4, stream>>>(qt, kt, foT, inp, w_proj,
                                      g3, b3, m3, v3, a3, (float*)d_out);
}
